// CrossModalFusion_17394617549515
// MI455X (gfx1250) — matrix operand order repaired and checked by reading, not compiled, not run on hardware
//
#include <hip/hip_runtime.h>
#include <hip/hip_bf16.h>
#include <math.h>

#define TT 2048
#define HH 512

typedef __bf16 bf16_t;
typedef bf16_t v16bf __attribute__((ext_vector_type(16)));
typedef float  v8f   __attribute__((ext_vector_type(8)));

static __device__ inline v8f zero8() {
  v8f z = {0.f, 0.f, 0.f, 0.f, 0.f, 0.f, 0.f, 0.f};
  return z;
}

static __device__ inline v8f wmma_bf16(v16bf a, v16bf b, v8f c) {
  // 8 args: (neg_a, A, neg_b, B, c_mod, C, reuse_a, reuse_b)
  return __builtin_amdgcn_wmma_f32_16x16x32_bf16(false, a, false, b, (short)0, c,
                                                 false, false);
}

// A fragment 16x32 bf16 from row-major [16 x lda] block at column k0.
// Lane L: m = L&15, h = L>>4. Element i -> K = 8h + i (i<8), 8h + 8 + i (i>=8).
// Per lane: two contiguous 8-element (16B) runs -> b128 loads.
static __device__ inline v16bf load_a_frag(const bf16_t* __restrict__ A, int lda,
                                           int k0, int lane) {
  const int m = lane & 15;
  const int h = lane >> 4;
  const bf16_t* p = A + (size_t)m * lda + k0 + 8 * h;
  v16bf a;
#pragma unroll
  for (int i = 0; i < 8; ++i) a[i] = p[i];
#pragma unroll
  for (int i = 8; i < 16; ++i) a[i] = p[i + 8];
  return a;
}

// B fragment (32x16 logical, contraction x columns) sourced from a ROW-MAJOR
// [N x ldk] matrix that is the TRANSPOSE of the logical B (contraction index
// contiguous in memory). Lane L: n = L&15, h = L>>4; element i -> contraction
// index k0 + 8h + i (i<8), k0 + 8h + 8 + i (i>=8). Two 16B runs per lane -> 2x b128.
static __device__ inline v16bf load_bt_frag(const bf16_t* __restrict__ Bt, int ldk,
                                            int n0, int k0, int lane) {
  const int n = n0 + (lane & 15);
  const int h = lane >> 4;
  const bf16_t* p = Bt + (size_t)n * ldk + k0 + 8 * h;
  v16bf b;
#pragma unroll
  for (int i = 0; i < 8; ++i) b[i] = p[i];
#pragma unroll
  for (int i = 8; i < 16; ++i) b[i] = p[i + 8];
  return b;
}

// ---------------- conversion kernels (f32 -> bf16) ---------------------------

// Activation pad: in [M x K] -> out [M x Kp], columns >= K zeroed.
__global__ __launch_bounds__(256) void cvt_colpad_kernel(
    const float* __restrict__ in, bf16_t* __restrict__ out, int M, int K, int Kp) {
  int idx = blockIdx.x * 256 + threadIdx.x;
  int total = M * Kp;
  if (idx >= total) return;
  int row = idx / Kp;
  int col = idx - row * Kp;
  out[idx] = (col < K) ? (bf16_t)in[(size_t)row * K + col] : (bf16_t)0.f;
}

// Weight transpose + pad: in [K x N] (row-major) -> out [N x Kp], k >= K zeroed.
__global__ __launch_bounds__(256) void cvt_wtrans_kernel(
    const float* __restrict__ in, bf16_t* __restrict__ out, int K, int Kp, int N) {
  int idx = blockIdx.x * 256 + threadIdx.x;
  int total = N * Kp;
  if (idx >= total) return;
  int n = idx / Kp;
  int k = idx - n * Kp;
  out[idx] = (k < K) ? (bf16_t)in[(size_t)k * N + n] : (bf16_t)0.f;
}

// ---------------- 16x(NT*16)-tile WMMA GEMM ----------------------------------
// One wave computes a 16 x (NT*16) output tile (A fragment reused NT times).
// A: [M x Kp] row-major bf16.  BT: [N x Kp] row-major bf16 (transposed weight).
// MODE 0: out = bf16(A@B + bias), row-major [M x N]
// MODE 1: out = f32(resid + A@B + bias), row-major [M x N]
// MODE 2: out = bf16(A@B + bias), transposed per batch: [B][N][TT]
// NT/MODE are compile-time: the inner loop is straight-line (no EXEC games,
// EXEC stays all-ones around every WMMA).
template <int NT, int MODE>
__global__ __launch_bounds__(256) void gemm_kernel(
    const bf16_t* __restrict__ A, const bf16_t* __restrict__ BT,
    const float* __restrict__ bias, const float* __restrict__ resid,
    bf16_t* __restrict__ outB, float* __restrict__ outF,
    int M, int N, int Kp, int n_base, int ngroups) {
  const int lane = threadIdx.x & 31;
  const int wave = threadIdx.x >> 5;
  const int gw = blockIdx.x * 8 + wave;
  const int mt = gw / ngroups;
  if (mt * 16 >= M) return;  // wave-uniform
  const int n0 = n_base + (gw - mt * ngroups) * (NT * 16);

  const bf16_t* Arow = A + (size_t)(mt * 16) * Kp;
  v8f c[NT];
#pragma unroll
  for (int t = 0; t < NT; ++t) c[t] = zero8();

  for (int k0 = 0; k0 < Kp; k0 += 32) {
    v16bf a = load_a_frag(Arow, Kp, k0, lane);
#pragma unroll
    for (int t = 0; t < NT; ++t) {
      v16bf b = load_bt_frag(BT, Kp, n0 + 16 * t, k0, lane);
      c[t] = wmma_bf16(a, b, c[t]);
    }
  }

  const int h = lane >> 4;
  const int nnl = lane & 15;
#pragma unroll
  for (int t = 0; t < NT; ++t) {
    const int n = n0 + t * 16 + nnl;
    const float bv = bias[n];
#pragma unroll
    for (int r = 0; r < 8; ++r) {
      const int m = mt * 16 + r + 8 * h;
      float v = c[t][r] + bv;
      if (MODE == 0) {
        outB[(size_t)m * N + n] = (bf16_t)v;
      } else if (MODE == 1) {
        outF[(size_t)m * N + n] = resid[(size_t)m * N + n] + v;
      } else {
        const int bb = m >> 11;        // m / TT
        const int tt = m & (TT - 1);   // m % TT
        outB[((size_t)bb * N + n) * (size_t)TT + tt] = (bf16_t)v;
      }
    }
  }
}

// ---------------- fused flash attention --------------------------------------
// One block (8 waves) per 16-query tile. Per 32-key step:
//   every wave: partial 16x32 scores over its private 64-wide H slice (4 WMMAs)
//   all 256 threads: deterministic fixed-order reduction of the 8 partials
//   wave0: row max/sum via half-row-per-lane + shfl_xor(16)
//   every wave: rebuild P fragment via exp on reduced scores, rescale its
//               private 16x64 O accumulators, P@V (4 WMMAs, V^T layout).
__global__ __launch_bounds__(256) void attn_kernel(
    const bf16_t* __restrict__ Q, const bf16_t* __restrict__ Km,
    const bf16_t* __restrict__ VTm, bf16_t* __restrict__ O) {
  __shared__ float S_part[8][16][32];  // per-wave scaled partial scores (16 KB)
  __shared__ float S_red[16][32];      // reduced scaled scores (2 KB)
  __shared__ float newm_s[16];
  __shared__ float alpha_s[16];
  __shared__ float l_s[16];

  const int tid = threadIdx.x;
  const int lane = tid & 31;
  const int wave = tid >> 5;
  const int rowblk = blockIdx.x * 16;  // global query row base
  const int batch = blockIdx.x >> 7;   // 128 query tiles per batch
  const bf16_t* Qrow = Q + (size_t)rowblk * HH;
  const bf16_t* Kb = Km + (size_t)batch * TT * HH;           // [T][H]
  const bf16_t* VTb = VTm + (size_t)batch * (size_t)HH * TT; // [H][T]

  const float scale = 0.044194173824159216f;  // 1/sqrt(512)

  float m_state = -1e30f, l_state = 0.f;  // maintained by all 32 lanes of wave0

  const int hbase = wave * 64;  // this wave's H slice
  const int h = lane >> 4;
  const int nn = lane & 15;

  // Q fragments are loop-invariant: hoist.
  v16bf qa0 = load_a_frag(Qrow, HH, hbase, lane);
  v16bf qa1 = load_a_frag(Qrow, HH, hbase + 32, lane);

  v8f o0 = zero8(), o1 = zero8(), o2 = zero8(), o3 = zero8();

  for (int j0 = 0; j0 < TT; j0 += 32) {
    // ---- prefetch next key block (global_prefetch_b8) ----
    if (j0 + 32 < TT) {
      __builtin_prefetch(Kb + (size_t)(j0 + 32 + lane) * HH + hbase, 0, 1);
      __builtin_prefetch(VTb + (size_t)(hbase + lane) * TT + j0 + 32, 0, 1);
      __builtin_prefetch(VTb + (size_t)(hbase + 32 + lane) * TT + j0 + 32, 0, 1);
    }

    // ---- partial scores over this wave's 64-wide H slice (4 WMMAs) ----
    v8f s0 = zero8(), s1 = zero8();
    {
      v16bf b00 = load_bt_frag(Kb, HH, j0, hbase, lane);
      v16bf b10 = load_bt_frag(Kb, HH, j0 + 16, hbase, lane);
      s0 = wmma_bf16(qa0, b00, s0);
      s1 = wmma_bf16(qa0, b10, s1);
      v16bf b01 = load_bt_frag(Kb, HH, j0, hbase + 32, lane);
      v16bf b11 = load_bt_frag(Kb, HH, j0 + 16, hbase + 32, lane);
      s0 = wmma_bf16(qa1, b01, s0);
      s1 = wmma_bf16(qa1, b11, s1);
    }
#pragma unroll
    for (int r = 0; r < 8; ++r) {
      S_part[wave][r + 8 * h][nn] = s0[r] * scale;
      S_part[wave][r + 8 * h][16 + nn] = s1[r] * scale;
    }
    __syncthreads();  // barrier A: partials visible

    // ---- parallel deterministic reduction: 2 elements per thread ----
#pragma unroll
    for (int e = tid; e < 512; e += 256) {
      float acc = 0.f;
#pragma unroll
      for (int w2 = 0; w2 < 8; ++w2) acc += (&S_part[w2][0][0])[e];
      (&S_red[0][0])[e] = acc;
    }
    __syncthreads();  // barrier B: reduced scores visible

    // ---- wave0: streaming softmax stats (half row per lane) ----
    if (wave == 0) {
      float vals[16];
      float vmax = -1e30f;
#pragma unroll
      for (int j = 0; j < 16; ++j) {
        vals[j] = S_red[nn][16 * h + j];
        vmax = fmaxf(vmax, vals[j]);
      }
      vmax = fmaxf(vmax, __shfl_xor(vmax, 16));
      float newm = fmaxf(m_state, vmax);
      float al = __expf(m_state - newm);
      float sum = 0.f;
#pragma unroll
      for (int j = 0; j < 16; ++j) sum += __expf(vals[j] - newm);
      sum += __shfl_xor(sum, 16);
      l_state = al * l_state + sum;
      m_state = newm;
      if (lane < 16) {
        newm_s[lane] = newm;
        alpha_s[lane] = al;
      }
    }
    __syncthreads();  // barrier C: newm/alpha visible

    // ---- P fragment in A layout, exp computed in parallel per wave ----
    const float nm = newm_s[nn];
    v16bf pa;
#pragma unroll
    for (int i = 0; i < 8; ++i)
      pa[i] = (bf16_t)__expf(S_red[nn][8 * h + i] - nm);
#pragma unroll
    for (int i = 8; i < 16; ++i)
      pa[i] = (bf16_t)__expf(S_red[nn][8 * h + 8 + i] - nm);

    // ---- rescale O accumulators by per-row alpha ----
#pragma unroll
    for (int r = 0; r < 8; ++r) {
      const float al = alpha_s[r + 8 * h];
      o0[r] *= al;
      o1[r] *= al;
      o2[r] *= al;
      o3[r] *= al;
    }

    // ---- O += P @ V over this wave's 64-wide H slice (4 WMMAs, V^T) ----
    {
      v16bf vb0 = load_bt_frag(VTb, TT, hbase + 0, j0, lane);
      v16bf vb1 = load_bt_frag(VTb, TT, hbase + 16, j0, lane);
      v16bf vb2 = load_bt_frag(VTb, TT, hbase + 32, j0, lane);
      v16bf vb3 = load_bt_frag(VTb, TT, hbase + 48, j0, lane);
      o0 = wmma_bf16(pa, vb0, o0);
      o1 = wmma_bf16(pa, vb1, o1);
      o2 = wmma_bf16(pa, vb2, o2);
      o3 = wmma_bf16(pa, vb3, o3);
    }
    // next-iteration barrier A orders S_part/S_red overwrites after all reads.
  }

  if (wave == 0 && lane < 16) l_s[lane] = l_state;
  __syncthreads();

#pragma unroll
  for (int r = 0; r < 8; ++r) {
    const int m = r + 8 * h;
    const float inv = 1.f / l_s[m];
    bf16_t* orow = O + (size_t)(rowblk + m) * HH + hbase + nn;
    orow[0] = (bf16_t)(o0[r] * inv);
    orow[16] = (bf16_t)(o1[r] * inv);
    orow[32] = (bf16_t)(o2[r] * inv);
    orow[48] = (bf16_t)(o3[r] * inv);
  }
}

// ---------------- launcher ---------------------------------------------------
extern "C" void kernel_launch(void* const* d_in, const int* in_sizes, int n_in,
                              void* d_out, int out_size, void* d_ws, size_t ws_size,
                              hipStream_t stream) {
  (void)in_sizes; (void)n_in; (void)out_size; (void)ws_size;
  const float* rgb = (const float*)d_in[0];
  const float* pose = (const float*)d_in[1];
  const float* Wq = (const float*)d_in[2];
  const float* bq = (const float*)d_in[3];
  const float* Wk = (const float*)d_in[4];
  const float* bk = (const float*)d_in[5];
  const float* Wv = (const float*)d_in[6];
  const float* bv = (const float*)d_in[7];
  const float* Wp = (const float*)d_in[8];
  const float* bp = (const float*)d_in[9];
  float* out = (float*)d_out;

  const int Bb = 8, T = 2048, DR = 400, DP = 256, H = 512;
  const int M = Bb * T;  // 16384
  const int DRp = 416;   // 400 padded to multiple of 32

  bf16_t* p = (bf16_t*)d_ws;
  bf16_t* rgbb = p;  p += (size_t)M * DRp;   // [M][DRp]
  bf16_t* poseb = p; p += (size_t)M * DP;    // [M][DP]
  bf16_t* wqT = p;   p += (size_t)H * DRp;   // [H][DRp]
  bf16_t* wkT = p;   p += (size_t)H * DP;    // [H][DP]
  bf16_t* wvT = p;   p += (size_t)H * DP;    // [H][DP]
  bf16_t* wpT = p;   p += (size_t)DR * H;    // [DR][H]
  bf16_t* Qb = p;    p += (size_t)M * H;     // [M][H]
  bf16_t* Kb = p;    p += (size_t)M * H;     // [M][H]
  bf16_t* VTb = p;   p += (size_t)M * H;     // [B][H][T]
  bf16_t* Ob = p;    p += (size_t)M * H;     // [M][H]

  auto g1 = [](int n) { return (n + 255) / 256; };

  cvt_colpad_kernel<<<g1(M * DRp), 256, 0, stream>>>(rgb, rgbb, M, DR, DRp);
  cvt_colpad_kernel<<<g1(M * DP), 256, 0, stream>>>(pose, poseb, M, DP, DP);
  cvt_wtrans_kernel<<<g1(H * DRp), 256, 0, stream>>>(Wq, wqT, DR, DRp, H);
  cvt_wtrans_kernel<<<g1(H * DP), 256, 0, stream>>>(Wk, wkT, DP, DP, H);
  cvt_wtrans_kernel<<<g1(H * DP), 256, 0, stream>>>(Wv, wvT, DP, DP, H);
  cvt_wtrans_kernel<<<g1(DR * H), 256, 0, stream>>>(Wp, wpT, H, H, DR);

  // projections: 16x64 tile per wave, 8 waves per block
  const int ngQ = H / 64;                 // 8 groups of 4 n-tiles
  const int wavesQ = (M / 16) * ngQ;      // 8192 -> 1024 blocks
  gemm_kernel<4, 0><<<(wavesQ + 7) / 8, 256, 0, stream>>>(
      rgbb, wqT, bq, nullptr, Qb, nullptr, M, H, DRp, 0, ngQ);
  gemm_kernel<4, 0><<<(wavesQ + 7) / 8, 256, 0, stream>>>(
      poseb, wkT, bk, nullptr, Kb, nullptr, M, H, DP, 0, ngQ);
  gemm_kernel<4, 2><<<(wavesQ + 7) / 8, 256, 0, stream>>>(
      poseb, wvT, bv, nullptr, VTb, nullptr, M, H, DP, 0, ngQ);  // V^T output

  attn_kernel<<<M / 16, 256, 0, stream>>>(Qb, Kb, VTb, Ob);

  // final projection: N = 400 = 6 full 64-wide groups + one 16-wide tail
  const int ngP = 6;                      // columns 0..383
  const int wavesP = (M / 16) * ngP;      // 6144 -> 768 blocks
  gemm_kernel<4, 1><<<(wavesP + 7) / 8, 256, 0, stream>>>(
      Ob, wpT, bp, rgb, nullptr, out, M, DR, H, 0, ngP);
  const int wavesTail = (M / 16);         // 1024 -> 128 blocks
  gemm_kernel<1, 1><<<(wavesTail + 7) / 8, 256, 0, stream>>>(
      Ob, wpT, bp, rgb, nullptr, out, M, DR, H, 384, 1);
}
